// DenseDilatedKnnGraph_66752381715110
// MI455X (gfx1250) — matrix operand order repaired and checked by reading, not compiled, not run on hardware
//
#include <hip/hip_runtime.h>
#include <hip/hip_fp16.h>

typedef __attribute__((ext_vector_type(16))) _Float16 v16h;
typedef __attribute__((ext_vector_type(8)))  float    v8f;
typedef unsigned int u32x4 __attribute__((ext_vector_type(4)));
typedef int          i32x4 __attribute__((ext_vector_type(4)));
typedef int          i32x8 __attribute__((ext_vector_type(8)));

#define B_    4
#define C_    64
#define N_    8192
#define KNN   9
#define DIL   2
#define KK    18                 // K_NEIGHBORS * DILATION
#define NT    (N_ / 16)          // 512 tiles per batch
#define CHUNK 256
#define NCHUNK (N_ / CHUNK)      // 32
#define TPC   (CHUNK / 16)       // 16 y-tiles per chunk
#define STRIDE (CHUNK + 4)       // LDS row stride
#define EPSN  1e-12f
#define FINF  3.0e38f
// bytes of packed B fragments per 16-col tile (2 kchunks * 32 lanes * 16 halves * 2B)
#define TILE_FRAG_BYTES 2048
#define CHUNK_FRAG_HALVES (TPC * 2 * 32 * 16)   // 16384 halves = 32KB

#if defined(__has_builtin)
#  if __has_builtin(__builtin_amdgcn_tensor_load_to_lds) && \
      __has_builtin(__builtin_amdgcn_s_wait_tensorcnt)
#    define HAVE_TDM 1
#  else
#    define HAVE_TDM 0
#  endif
#else
#  define HAVE_TDM 0
#endif

// ---------------------------------------------------------------------------
// Pass 1: L2-normalize along C and scatter into WMMA fragment layout (f16).
// mode 0: A-matrix 16x32 layout from x; mode 1: B-matrix 32x16 layout from y
// plus ||y||^2. Storage: frag[(((b*NT+tile)*2+kchunk)*32+lane)*16+h] so each
// lane's v16h operand is one contiguous 32-byte load.
// ---------------------------------------------------------------------------
__global__ void pack_norm_kernel(const float* __restrict__ src,
                                 _Float16* __restrict__ frag,
                                 float* __restrict__ ysq, int mode) {
  int pid = blockIdx.x * blockDim.x + threadIdx.x;
  if (pid >= B_ * N_) return;
  int b = pid / N_;
  int n = pid % N_;

  float vals[C_];
  float ss = 0.f;
  const float* p = src + (size_t)b * C_ * N_ + n;
#pragma unroll
  for (int c = 0; c < C_; ++c) {
    float v = p[(size_t)c * N_];
    vals[c] = v;
    ss += v * v;
  }
  float inv = 1.0f / fmaxf(sqrtf(ss), EPSN);
  if (mode == 1) ysq[pid] = ss * inv * inv;

  int tile = n >> 4;
  int r    = n & 15;
  size_t tbase = (size_t)(b * NT + tile) * 2;
#pragma unroll
  for (int c = 0; c < C_; ++c) {
    int kc = c >> 5;
    int kk = c & 31;
    int lane, h;
    if (mode == 0) {            // A 16x32 f16 per-lane layout (ISA 7.12.2)
      int hi = (kk >> 3) & 1;
      lane = r + 16 * hi;
      if      (kk < 8)  h = kk;
      else if (kk < 16) h = kk - 8;
      else if (kk < 24) h = kk - 8;
      else              h = kk - 16;
    } else {                    // B 32x16: col = lane%16, same K order as A
      int hi = (kk >> 3) & 1;
      lane = r + 16 * hi;
      h = (kk & 7) + ((kk >> 4) << 3);
    }
    frag[((tbase + kc) * 32 + lane) * 16 + h] = (_Float16)(vals[c] * inv);
  }
}

// ---------------------------------------------------------------------------
// Pass 2: fused WMMA distance GEMM + register-resident streaming top-18.
// One workgroup (4 waves) per 16 x-rows. TDM stages each chunk's B fragments
// into LDS, overlapped with the selection phase of the previous chunk.
// ---------------------------------------------------------------------------
__global__ __launch_bounds__(128) void knn_wmma_kernel(
    const _Float16* __restrict__ Afrag,
    const _Float16* __restrict__ Bfrag,
    const float* __restrict__ ysq,
    int* __restrict__ out) {
  __shared__ float lds_dist[16][STRIDE];
#if HAVE_TDM
  __shared__ __attribute__((aligned(32))) _Float16 Bbuf[CHUNK_FRAG_HALVES];
#endif

  const int tid   = threadIdx.x;
  const int lane  = tid & 31;
  const int wid   = tid >> 5;           // wave 0..3
  const int bid   = blockIdx.x;         // b*NT + xtile
  const int b     = bid / NT;
  const int xtile = bid % NT;

  // A fragments (both K-chunks) for this row tile; shared by all 4 waves.
  const v16h* ap = (const v16h*)(Afrag + ((size_t)bid * 2 * 32 + lane) * 16);
  v16h a0 = ap[0];
  v16h a1 = ap[32];

  // Per-row sorted top-18 lists: lane j (< KK) holds entry j of each row's
  // ascending list. 4 rows per wave (rows wid*4 .. wid*4+3).
  float lvr[4];
  int   lir[4];
#pragma unroll
  for (int rl = 0; rl < 4; ++rl) { lvr[rl] = FINF; lir[rl] = -1; }

#if HAVE_TDM
  // D# builder for one chunk's contiguous 32KB fragment block.
  const unsigned bbuf_off = (unsigned)(size_t)(void*)&Bbuf[0]; // LDS byte offset
  auto issue_tdm = [&](int ch) {
    unsigned long long gaddr =
        (unsigned long long)(size_t)(Bfrag) +
        ((unsigned long long)(b * NT + ch * TPC)) * TILE_FRAG_BYTES;
    u32x4 g0;
    g0.x = 1u;                                        // count=1 (valid user D#)
    g0.y = bbuf_off;                                  // lds_addr
    g0.z = (unsigned)(gaddr & 0xFFFFFFFFull);         // global_addr[31:0]
    g0.w = (unsigned)((gaddr >> 32) & 0x1FFFFFFull)   // global_addr[56:32]
         | (2u << 30);                                // type = 2 ("image")
    i32x8 g1;
    g1[0] = 3 << 16;               // data_size = 3 (8 bytes)
    g1[1] = (int)(4096u << 16);    // tensor_dim0[15:0] = 4096 (8B units)
    g1[2] = 1 << 16;               // tensor_dim0 hi = 0, tensor_dim1 = 1
    g1[3] = (int)(4096u << 16);    // tile_dim0 = 4096
    g1[4] = 1;                     // tile_dim1 = 1, tile_dim2 = 0
    g1[5] = 4096;                  // tensor_dim0_stride = 4096
    g1[6] = 0;
    g1[7] = 0;
    i32x4 z4 = {};
    i32x8 z8 = {};
    __builtin_amdgcn_tensor_load_to_lds(g0, g1, z4, z4, z8, 0);
  };
  if (wid == 0) {
    issue_tdm(0);
    __builtin_amdgcn_s_wait_tensorcnt(0);
  }
  __syncthreads();
#endif

  for (int ch = 0; ch < NCHUNK; ++ch) {
    // ---- GEMM phase: 16 rows x CHUNK cols of distances into LDS ----
    for (int t = wid; t < TPC; t += 4) {
#if HAVE_TDM
      const v16h* bp = (const v16h*)&Bbuf[((t * 2) * 32 + lane) * 16];
#else
      const v16h* bp = (const v16h*)(Bfrag +
          ((size_t)(b * NT + ch * TPC + t) * 2 * 32 + lane) * 16);
#endif
      v16h b0 = bp[0];
      v16h b1 = bp[32];
      v8f c = {};
      c = __builtin_amdgcn_wmma_f32_16x16x32_f16(false, a0, false, b0,
                                                 (short)0, c, false, false);
      c = __builtin_amdgcn_wmma_f32_16x16x32_f16(false, a1, false, b1,
                                                 (short)0, c, false, false);
      int coll  = t * 16 + (lane & 15);
      float yq  = ysq[b * N_ + ch * CHUNK + coll];
      int rbase = (lane >> 4) * 8;
#pragma unroll
      for (int v = 0; v < 8; ++v)
        lds_dist[rbase + v][coll] = yq - 2.0f * c[v];
    }
    __syncthreads();

#if HAVE_TDM
    if (wid == 0 && ch + 1 < NCHUNK) issue_tdm(ch + 1);  // overlap w/ selection
#endif

    // ---- Selection phase: ballot-filtered sorted insert, regs only ----
#pragma unroll
    for (int rl = 0; rl < 4; ++rl) {
      int r = wid * 4 + rl;
      float lv = lvr[rl];
      int   li = lir[rl];
      float thr = __shfl(lv, KK - 1, 32);    // current 18th-best
      for (int p0 = 0; p0 < CHUNK; p0 += 32) {
        float v = lds_dist[r][p0 + lane];
        unsigned mask = (unsigned)__ballot(v < thr);
        while (mask) {
          int s = __builtin_ctz(mask);
          mask &= mask - 1;
          float cv = __shfl(v, s, 32);
          int   ci = ch * CHUNK + p0 + s;
          // sorted insert of (cv,ci) into ascending list held in lanes 0..17
          float uv = __shfl_up(lv, 1, 32);
          int   ui = __shfl_up(li, 1, 32);
          if (lv <= cv) {
            /* keep */
          } else if (lane == 0 || uv <= cv) {
            lv = cv; li = ci;
          } else {
            lv = uv; li = ui;
          }
        }
        thr = __shfl(lv, KK - 1, 32);
      }
      lvr[rl] = lv;
      lir[rl] = li;
    }

#if HAVE_TDM
    if (wid == 0) __builtin_amdgcn_s_wait_tensorcnt(0);
#endif
    __syncthreads();
  }

  // ---- Emit edge_index[:, b, n, ::2] : list entry 2j lives in lane 2j ----
#pragma unroll
  for (int rl = 0; rl < 4; ++rl) {
    int r = wid * 4 + rl;
    int n = xtile * 16 + r;
    int nn = __shfl(lir[rl], 2 * lane, 32);
    if (lane < KNN) {
      out[(((size_t)0 * B_ + b) * N_ + n) * KNN + lane] = nn;  // neighbors
      out[(((size_t)1 * B_ + b) * N_ + n) * KNN + lane] = n;   // centers
    }
  }
}

// ---------------------------------------------------------------------------
extern "C" void kernel_launch(void* const* d_in, const int* in_sizes, int n_in,
                              void* d_out, int out_size, void* d_ws, size_t ws_size,
                              hipStream_t stream) {
  const float* x = (const float*)d_in[0];
  const float* y = (const float*)d_in[1];

  const size_t fragBytes = (size_t)B_ * N_ * C_ * sizeof(_Float16);  // 4 MB each
  _Float16* Afrag = (_Float16*)d_ws;
  _Float16* Bfrag = (_Float16*)((char*)d_ws + fragBytes);
  float*    ysq   = (float*)((char*)d_ws + 2 * fragBytes);           // 128 KB

  int npts = B_ * N_;
  pack_norm_kernel<<<(npts + 255) / 256, 256, 0, stream>>>(x, Afrag, ysq, 0);
  pack_norm_kernel<<<(npts + 255) / 256, 256, 0, stream>>>(y, Bfrag, ysq, 1);
  knn_wmma_kernel<<<B_ * NT, 128, 0, stream>>>(Afrag, Bfrag, ysq, (int*)d_out);
}
